// MoeLayer_87342454932202
// MI455X (gfx1250) — compile-verified, matrix operand order checked by reading, not run on hardware
//
#include <hip/hip_runtime.h>
#include <hip/hip_bf16.h>
#include <math.h>

// Problem sizes (fixed by the reference)
#define BB   4
#define SS   2048
#define HH   1024
#define MM   4096
#define EE   8
#define TOK  (BB * SS)          // 8192 tokens

// GEMM tiling
#define BM   64                 // rows (tokens) per block
#define BN   256                // cols per block
#define BK   32                 // K-step
#define LDP  40                 // padded LDS row stride (ushorts): 80B = 20 banks,
                                // r*20 mod 64 distinct for r=0..15 -> conflict-free

typedef __attribute__((ext_vector_type(16))) __bf16 v16bf;
typedef __attribute__((ext_vector_type(8)))  float  v8f;

// ---------------------------------------------------------------- helpers ---
static __device__ __forceinline__ unsigned short f32_to_bf16(float f) {
  unsigned int u = __float_as_uint(f);
  unsigned int r = (u + 0x7FFFu + ((u >> 16) & 1u)) >> 16;   // RNE
  return (unsigned short)r;
}

// Load one 16x32 bf16 fragment from an LDS subtile stored [16 rows][LDP]
// (k contiguous), matching the CDNA5 16-bit A/B matrix VGPR layout:
//   lanes 0-15 : row = lane,    VGPR p holds K = {2p, 2p+1}        (p<4)
//                                          K = {16+2(p-4), ...}    (p>=4)
//   lanes 16-31: row = lane-16, same pattern with K += 8
static __device__ __forceinline__ v16bf load_frag(
    const unsigned short* __restrict__ tile, int lane) {
  union { v16bf v; unsigned int u[8]; } f;
  const int r  = lane & 15;
  const int kh = (lane >> 4) << 3;                 // 0 or 8
  const unsigned int* p = (const unsigned int*)(tile + r * LDP);
  #pragma unroll
  for (int q = 0; q < 8; ++q) {
    int kb = ((q < 4) ? (q << 1) : (16 + ((q - 4) << 1))) + kh;
    f.u[q] = p[kb >> 1];                           // packed bf16 pair (k,k+1)
  }
  return f.v;
}

// ---------------------------------------------------------------- kernels ---
__global__ __launch_bounds__(256) void cvt_f32_bf16(
    const float* __restrict__ s, unsigned short* __restrict__ d, int n) {
  for (int i = blockIdx.x * blockDim.x + threadIdx.x; i < n;
       i += gridDim.x * blockDim.x)
    d[i] = f32_to_bf16(s[i]);
}

__global__ __launch_bounds__(256) void zero_f32(float* __restrict__ d, int n) {
  for (int i = blockIdx.x * blockDim.x + threadIdx.x; i < n;
       i += gridDim.x * blockDim.x)
    d[i] = 0.0f;
}

// Router: one wave32 per token. logits -> softmax(E=8) -> top2 -> renorm.
__global__ __launch_bounds__(256) void router_kernel(
    const float* __restrict__ x, const float* __restrict__ Wr,
    float* __restrict__ wts) {
  const int lane = threadIdx.x & 31;
  const int warp = threadIdx.x >> 5;
  const int t = blockIdx.x * 8 + warp;
  if (t >= TOK) return;
  const float* xt = x + (size_t)t * HH;
  float acc[EE];
  #pragma unroll
  for (int e = 0; e < EE; ++e) acc[e] = 0.0f;
  for (int h = lane; h < HH; h += 32) {
    float xv = xt[h];
    #pragma unroll
    for (int e = 0; e < EE; ++e) acc[e] += xv * Wr[e * HH + h];
  }
  #pragma unroll
  for (int e = 0; e < EE; ++e)
    #pragma unroll
    for (int off = 16; off > 0; off >>= 1)
      acc[e] += __shfl_down(acc[e], off, 32);
  if (lane == 0) {
    float mx = acc[0];
    #pragma unroll
    for (int e = 1; e < EE; ++e) mx = fmaxf(mx, acc[e]);
    float p[EE], s = 0.0f;
    #pragma unroll
    for (int e = 0; e < EE; ++e) { p[e] = expf(acc[e] - mx); s += p[e]; }
    #pragma unroll
    for (int e = 0; e < EE; ++e) p[e] /= s;
    int i1 = 0;
    #pragma unroll
    for (int e = 1; e < EE; ++e) if (p[e] > p[i1]) i1 = e;
    int i2 = (i1 == 0) ? 1 : 0;
    #pragma unroll
    for (int e = 0; e < EE; ++e)
      if (e != i1 && p[e] > p[i2]) i2 = e;
    float sn = p[i1] + p[i2];
    #pragma unroll
    for (int e = 0; e < EE; ++e)
      wts[(size_t)t * EE + e] =
          (e == i1) ? (p[i1] / sn) : ((e == i2) ? (p[i2] / sn) : 0.0f);
  }
}

// Tiled WMMA GEMM: C[TOK x Ntot] = A[TOK x Kdim] * Bw[Ntot x Kdim]^T
//   EPI==0: C = exact-gelu(C + bias)          -> bf16 Hout
//   EPI==1: Out += (C + bias) * wts[row][e]   -> f32 accumulate
// Block tile 64(M) x 256(N), K-step 32; 8 waves (2x4), wave tile 32x64
// = 2x4 accumulators of v8f -> 8 WMMA per K-step per wave.
template <int EPI>
__global__ __launch_bounds__(256) void moe_gemm(
    const unsigned short* __restrict__ A,    // bf16 [TOK][Kdim]
    const unsigned short* __restrict__ Bw,   // bf16 [Ntot][Kdim]
    const float* __restrict__ bias,          // [Ntot]
    int Kdim, int Ntot,
    unsigned short* __restrict__ Hout,       // EPI==0
    const float* __restrict__ wts, int expert,
    float* __restrict__ Out) {               // EPI==1
  __shared__ unsigned short sA[BM * LDP];    //  5 KiB
  __shared__ unsigned short sB[BN * LDP];    // 20 KiB

  const int tid  = threadIdx.x;
  const int lane = tid & 31;
  const int wave = tid >> 5;
  const int wm   = wave >> 2;                  // 0..1 : 32-row band
  const int wn   = wave & 3;                   // 0..3 : 64-col band
  const int rowblk = blockIdx.y * BM;
  const int colblk = blockIdx.x * BN;

  // cooperative tile-load assignments (128-bit coalesced, K contiguous)
  const int ar = tid >> 2, aks = (tid & 3) << 3;    // A: 64 rows x 32 k
  const int br = tid;                               // B: 256 rows x 32 k
  const unsigned short* aptr = A  + (size_t)(rowblk + ar) * Kdim + aks;
  const unsigned short* bptr = Bw + (size_t)(colblk + br) * Kdim;

  v8f acc[2][4] = {};

  for (int k0 = 0; k0 < Kdim; k0 += BK) {
    uint4 av = *(const uint4*)(aptr + k0);
    uint4 bv[4];
    #pragma unroll
    for (int q = 0; q < 4; ++q)
      bv[q] = *(const uint4*)(bptr + k0 + q * 8);
    if (k0 + BK < Kdim) {                      // hint next K tile toward L2
      __builtin_prefetch(aptr + k0 + BK, 0, 1);
      __builtin_prefetch(bptr + k0 + BK, 0, 1);
    }
    __syncthreads();
    *(uint4*)(&sA[ar * LDP + aks]) = av;
    #pragma unroll
    for (int q = 0; q < 4; ++q)
      *(uint4*)(&sB[br * LDP + q * 8]) = bv[q];
    __syncthreads();

    v16bf afr[2], bfr[4];
    #pragma unroll
    for (int mi = 0; mi < 2; ++mi)
      afr[mi] = load_frag(&sA[(wm * 32 + mi * 16) * LDP], lane);
    #pragma unroll
    for (int ni = 0; ni < 4; ++ni)
      bfr[ni] = load_frag(&sB[(wn * 64 + ni * 16) * LDP], lane);

    #pragma unroll
    for (int mi = 0; mi < 2; ++mi)
      #pragma unroll
      for (int ni = 0; ni < 4; ++ni)
        acc[mi][ni] = __builtin_amdgcn_wmma_f32_16x16x32_bf16(
            false, afr[mi], false, bfr[ni], (short)0, acc[mi][ni],
            false, false);
  }

  // C/D layout: VGPR r -> M=r (lanes 0-15) / M=r+8 (lanes 16-31); N = lane&15
  const int nL   = lane & 15;
  const int mOff = (lane >> 4) << 3;
  #pragma unroll
  for (int mi = 0; mi < 2; ++mi) {
    #pragma unroll
    for (int ni = 0; ni < 4; ++ni) {
      const int col = colblk + wn * 64 + ni * 16 + nL;
      const float bv = bias[col];
      #pragma unroll
      for (int r = 0; r < 8; ++r) {
        const int row = rowblk + wm * 32 + mi * 16 + mOff + r;
        float v = acc[mi][ni][r] + bv;
        if (EPI == 0) {
          v = 0.5f * v * (1.0f + erff(v * 0.70710678118654752f));  // exact gelu
          Hout[(size_t)row * Ntot + col] = f32_to_bf16(v);
        } else {
          const float w = wts[(size_t)row * EE + expert];
          Out[(size_t)row * Ntot + col] += v * w;
        }
      }
    }
  }
}

// ----------------------------------------------------------------- launch ---
extern "C" void kernel_launch(void* const* d_in, const int* in_sizes, int n_in,
                              void* d_out, int out_size, void* d_ws,
                              size_t ws_size, hipStream_t stream) {
  (void)in_sizes; (void)n_in; (void)out_size; (void)ws_size;
  const float* x  = (const float*)d_in[0];   // [B,S,H]
  const float* Wr = (const float*)d_in[1];   // [E,H]
  const float* W1 = (const float*)d_in[2];   // [E,M,H]
  const float* b1 = (const float*)d_in[3];   // [E,M]
  const float* W2 = (const float*)d_in[4];   // [E,H,M]
  const float* b2 = (const float*)d_in[5];   // [E,H]
  float* out = (float*)d_out;                // [B,S,H]

  // workspace carve (bytes):  xh 16MiB | w1h 64MiB | w2h 64MiB | h 64MiB | wts
  char* w = (char*)d_ws;
  unsigned short* xh  = (unsigned short*)(w);
  unsigned short* w1h = (unsigned short*)(w + (16ull  << 20));
  unsigned short* w2h = (unsigned short*)(w + (80ull  << 20));
  unsigned short* hb  = (unsigned short*)(w + (144ull << 20));
  float*          wts = (float*)        (w + (208ull << 20));

  cvt_f32_bf16<<<4096, 256, 0, stream>>>(x,  xh,  TOK * HH);
  cvt_f32_bf16<<<8192, 256, 0, stream>>>(W1, w1h, EE * MM * HH);
  cvt_f32_bf16<<<8192, 256, 0, stream>>>(W2, w2h, EE * HH * MM);
  router_kernel<<<TOK / 8, 256, 0, stream>>>(x, Wr, wts);
  zero_f32<<<4096, 256, 0, stream>>>(out, TOK * HH);

  for (int e = 0; e < EE; ++e) {
    // GEMM1: h = gelu(x @ W1[e]^T + b1[e])   -> hb (bf16 [TOK][MM])
    moe_gemm<0><<<dim3(MM / BN, TOK / BM), 256, 0, stream>>>(
        xh, w1h + (size_t)e * MM * HH, b1 + (size_t)e * MM,
        HH, MM, hb, nullptr, 0, nullptr);
    // GEMM2: out += (h @ W2[e]^T + b2[e]) * wts[:,e]
    moe_gemm<1><<<dim3(HH / BN, TOK / BM), 256, 0, stream>>>(
        hb, w2h + (size_t)e * HH * MM, b2 + (size_t)e * HH,
        MM, HH, nullptr, wts, e, out);
  }
}
